// RetNetWL_644245094447
// MI455X (gfx1250) — matrix operand order repaired and checked by reading, not compiled, not run on hardware
//
#include <hip/hip_runtime.h>
#include <hip/hip_bf16.h>
#include <math.h>

// ---------------- problem constants ----------------
constexpr int B_  = 16;
constexpr int T_  = 8192;
constexpr int D_  = 128;
constexpr int L_  = 4;
constexpr int TOK_ = B_ * T_;          // 131072 tokens
constexpr int NC_ = 32;                // EMA scan chunks
constexpr int CL_ = T_ / NC_;          // 256 steps per chunk

typedef __attribute__((ext_vector_type(16))) _Float16 v16h;
typedef __attribute__((ext_vector_type(8)))  _Float16 v8h;
typedef __attribute__((ext_vector_type(8)))  float    v8f;

__device__ __forceinline__ float gelu_exact(float x) {
    return 0.5f * x * (1.0f + erff(x * 0.70710678118654752f));
}

__device__ __forceinline__ float wave_red(float v) {
    #pragma unroll
    for (int o = 16; o > 0; o >>= 1) v += __shfl_xor(v, o, 32);
    return v;
}

__device__ __forceinline__ v16h mk16(v8h lo, v8h hi) {
    v16h r;
    #pragma unroll
    for (int i = 0; i < 8; ++i) { r[i] = lo[i]; r[i + 8] = hi[i]; }
    return r;
}

// ---------------- weight transpose + f32->f16 : dst[N][K] = src[K][N] ----------------
__global__ __launch_bounds__(256) void k_transpose_f16(const float* __restrict__ src,
                                                       _Float16* __restrict__ dst,
                                                       int K, int N) {
    int tid = blockIdx.x * 256 + threadIdx.x;
    if (tid >= K * N) return;
    int k = tid / N, n = tid % N;
    dst[(size_t)n * K + k] = (_Float16)src[(size_t)k * N + n];
}

// ---------------- embed: h = seq @ W[2,128] + b ----------------
__global__ __launch_bounds__(256) void k_embed(const float* __restrict__ seq,
                                               const float* __restrict__ ew,
                                               const float* __restrict__ eb,
                                               float* __restrict__ h) {
    size_t tid = (size_t)blockIdx.x * 256 + threadIdx.x;   // over TOK_*D_
    size_t tok = tid >> 7;
    int d = (int)(tid & 127);
    float s0 = seq[tok * 2 + 0], s1 = seq[tok * 2 + 1];
    h[tid] = s0 * ew[d] + s1 * ew[128 + d] + eb[d];
}

// ---------------- LayerNorm over D=128, one wave per token ----------------
// W16: additionally emit an f16 copy (GEMM A operand)
template <bool W16>
__global__ __launch_bounds__(256) void k_ln(const float* __restrict__ h,
                                            const float* __restrict__ g,
                                            const float* __restrict__ bb,
                                            float* __restrict__ xn,
                                            _Float16* __restrict__ xn16) {
    int wave = threadIdx.x >> 5, lane = threadIdx.x & 31;
    size_t tok = (size_t)blockIdx.x * 8 + wave;
    size_t base = tok * 128 + lane;
    float x[4];
    #pragma unroll
    for (int i = 0; i < 4; ++i) x[i] = h[base + 32 * i];
    float s = wave_red(x[0] + x[1] + x[2] + x[3]);
    float mean = s * (1.0f / 128.0f);
    float sq = 0.f;
    #pragma unroll
    for (int i = 0; i < 4; ++i) { float t = x[i] - mean; sq += t * t; }
    float var = wave_red(sq) * (1.0f / 128.0f);
    float rs = rsqrtf(var + 1e-5f);
    #pragma unroll
    for (int i = 0; i < 4; ++i) {
        int d = lane + 32 * i;
        float v = (x[i] - mean) * rs * g[d] + bb[d];
        xn[base + 32 * i] = v;
        if constexpr (W16) xn16[base + 32 * i] = (_Float16)v;
    }
}

// ---------------- EMA blocked scan (3 phases) ----------------
// phase A: local scan per (b, chunk, d); stores local states + chunk carries
__global__ __launch_bounds__(256) void k_ema_a(const float* __restrict__ xn,
                                               const float* __restrict__ logit,
                                               float* __restrict__ emaf,
                                               float* __restrict__ carries) {
    int tid = blockIdx.x * 256 + threadIdx.x;       // B*NC*D = 65536
    int d = tid % D_;
    int c = (tid / D_) % NC_;
    int b = tid / (D_ * NC_);
    float a = 1.0f / (1.0f + __expf(-logit[d]));
    float om = 1.0f - a;
    size_t base = ((size_t)b * T_ + (size_t)c * CL_) * D_ + d;
    float s = 0.f;
    for (int i = 0; i < CL_; ++i) {
        s = a * s + om * xn[base + (size_t)i * D_];
        emaf[base + (size_t)i * D_] = s;
    }
    carries[((size_t)b * NC_ + c) * D_ + d] = s;
}
// phase B: serial scan of 32 chunk carries -> incoming state per chunk
__global__ __launch_bounds__(256) void k_ema_b(const float* __restrict__ carries,
                                               const float* __restrict__ logit,
                                               float* __restrict__ Sin) {
    int tid = blockIdx.x * 256 + threadIdx.x;       // B*D = 2048
    int d = tid % D_, b = tid / D_;
    float a = 1.0f / (1.0f + __expf(-logit[d]));
    float aC = a;
    #pragma unroll
    for (int i = 0; i < 8; ++i) aC *= aC;           // a^256
    float S = 0.f;
    for (int c = 0; c < NC_; ++c) {
        Sin[((size_t)b * NC_ + c) * D_ + d] = S;
        S = carries[((size_t)b * NC_ + c) * D_ + d] + aC * S;
    }
}
// phase C: finalize: ema16 = f16(local + a^{i+1} * S_in)
__global__ __launch_bounds__(256) void k_ema_c(const float* __restrict__ emaf,
                                               const float* __restrict__ logit,
                                               const float* __restrict__ Sin,
                                               _Float16* __restrict__ ema16) {
    int tid = blockIdx.x * 256 + threadIdx.x;
    int d = tid % D_;
    int c = (tid / D_) % NC_;
    int b = tid / (D_ * NC_);
    float a = 1.0f / (1.0f + __expf(-logit[d]));
    float S = Sin[((size_t)b * NC_ + c) * D_ + d];
    size_t base = ((size_t)b * T_ + (size_t)c * CL_) * D_ + d;
    float p = a;
    for (int i = 0; i < CL_; ++i) {
        float v = emaf[base + (size_t)i * D_] + p * S;
        ema16[base + (size_t)i * D_] = (_Float16)v;
        p *= a;
    }
}

// ---------------- WMMA GEMM, double-buffered LDS, f16 A and B ----------------
// A: [tokens][KD] f16, Bt: [NN][KD] f16 (pre-transposed weight)
// EPI 0: h += A@W + bias + dwconv(xn)     (NN=128)
// EPI 1: Out16 = f16(gelu(A@W + bias))    (NN=256)
// EPI 2: h += A@W + bias                  (NN=128)
template <int KD, int NN, int EPI>
__global__ __launch_bounds__(256) void k_gemm(const _Float16* __restrict__ Ag,
                                              const _Float16* __restrict__ Bt,
                                              float* __restrict__ Out,
                                              _Float16* __restrict__ Out16,
                                              const float* __restrict__ bias,
                                              const float* __restrict__ xnC,
                                              const float* __restrict__ dww,
                                              const float* __restrict__ dwb) {
    constexpr int KC = 64;             // K chunk staged in LDS
    constexpr int SL = KC + 8;         // padded LDS row stride (halves), 16B aligned
    constexpr int NHALF = NN / 2;
    constexpr int CT = NHALF / 16;     // 16x16 col tiles per wave
    constexpr int AG = 2;              // A v8h groups per thread (64*KC/8/256)
    constexpr int BG = NN / 32;        // B v8h groups per thread (NN*KC/8/256)

    __shared__ __align__(16) _Float16 Als[2][64 * SL];
    __shared__ __align__(16) _Float16 Bls[2][NN * SL];

    const int tid = threadIdx.x;
    const int wv = tid >> 5, lane = tid & 31;
    const int rs = wv & 3;             // row stripe (16 rows)
    const int ch = wv >> 2;            // column half
    const int lrow = lane & 15;
    const int khi = lane >> 4;
    const int tok0 = blockIdx.x * 64;

    v8h sa[AG], sb[BG];                // register staging for next chunk

    auto gload = [&](int kc) {
        #pragma unroll
        for (int j = 0; j < AG; ++j) {
            int gi = tid + 256 * j;            // over 512 groups of 8 halves
            int r = gi >> 3, kg = gi & 7;
            sa[j] = *(const v8h*)(Ag + (size_t)(tok0 + r) * KD + kc + kg * 8);
        }
        #pragma unroll
        for (int j = 0; j < BG; ++j) {
            int gi = tid + 256 * j;            // over NN*8 groups
            int n = gi >> 3, kg = gi & 7;
            sb[j] = *(const v8h*)(Bt + (size_t)n * KD + kc + kg * 8);
        }
    };
    auto lstore = [&](int buf) {
        #pragma unroll
        for (int j = 0; j < AG; ++j) {
            int gi = tid + 256 * j;
            int r = gi >> 3, kg = gi & 7;
            *(v8h*)(&Als[buf][r * SL + kg * 8]) = sa[j];
        }
        #pragma unroll
        for (int j = 0; j < BG; ++j) {
            int gi = tid + 256 * j;
            int n = gi >> 3, kg = gi & 7;
            *(v8h*)(&Bls[buf][n * SL + kg * 8]) = sb[j];
        }
    };

    v8f acc[CT];
    #pragma unroll
    for (int c = 0; c < CT; ++c) {
        v8f z = {0.f, 0.f, 0.f, 0.f, 0.f, 0.f, 0.f, 0.f};
        acc[c] = z;
    }

    // prefetch next M-tile's A stream into L2 while we work on this one
    if (tok0 + 128 <= TOK_)
        __builtin_prefetch(Ag + (size_t)(tok0 + 64) * KD, 0, 1);

    gload(0);
    lstore(0);
    int buf = 0;
    for (int kc = 0; kc < KD; kc += KC) {
        __syncthreads();                       // stores into `buf` now visible
        const bool hasNext = (kc + KC) < KD;
        if (hasNext) gload(kc + KC);           // global loads in flight during compute

        const _Float16* Ab = &Als[buf][(rs * 16 + lrow) * SL];
        #pragma unroll
        for (int k0 = 0; k0 < KC; k0 += 32) {
            // A fragment: ISA 16-bit A layout (K interleave 0-7/16-23 vs 8-15/24-31)
            v8h alo = *(const v8h*)(Ab + k0 + khi * 8);
            v8h ahi = *(const v8h*)(Ab + k0 + 16 + khi * 8);
            v16h af = mk16(alo, ahi);
            #pragma unroll
            for (int c = 0; c < CT; ++c) {
                const _Float16* brow = &Bls[buf][(ch * NHALF + c * 16 + lrow) * SL];
                // B fragment: same K interleave as A (0-7/16-23 vs 8-15/24-31)
                v8h blo = *(const v8h*)(brow + k0 + khi * 8);
                v8h bhi = *(const v8h*)(brow + k0 + 16 + khi * 8);
                v16h bf = mk16(blo, bhi);
                acc[c] = __builtin_amdgcn_wmma_f32_16x16x32_f16(
                    false, af, false, bf, (short)0, acc[c], false, false);
            }
        }
        if (hasNext) lstore(buf ^ 1);          // waits loads, fills other buffer
        buf ^= 1;
    }

    // epilogue: C layout — VGPR r: lanes0-15 M=r, lanes16-31 M=8+r; N=lane&15
    #pragma unroll
    for (int c = 0; c < CT; ++c) {
        int col = ch * NHALF + c * 16 + lrow;
        #pragma unroll
        for (int r = 0; r < 8; ++r) {
            int m = rs * 16 + r + khi * 8;
            size_t row = (size_t)tok0 + m;
            float v = acc[c][r] + bias[col];
            if constexpr (EPI == 0) {
                int t = (int)(row & (T_ - 1));
                float conv = dwb[col];
                #pragma unroll
                for (int j = 0; j < 5; ++j) {
                    int tt = t + j - 2;
                    if (tt >= 0 && tt < T_)
                        conv += xnC[(row + (size_t)(j - 2)) * 128 + col] * dww[col * 5 + j];
                }
                Out[row * 128 + col] += v + conv;
            } else if constexpr (EPI == 1) {
                Out16[row * NN + col] = (_Float16)gelu_exact(v);
            } else {
                Out[row * 128 + col] += v;
            }
        }
    }
}

// ---------------- masked mean pool over T ----------------
__global__ __launch_bounds__(128) void k_pool(const float* __restrict__ h,
                                              const unsigned char* __restrict__ mask,
                                              float* __restrict__ pooled) {
    int b = blockIdx.x, d = threadIdx.x;
    float sum = 0.f, cnt = 0.f;
    for (int t = 0; t < T_; ++t) {
        size_t idx = (size_t)b * T_ + t;
        float valid = mask[idx] ? 0.f : 1.f;
        sum += h[idx * 128 + d] * valid;
        cnt += valid;
    }
    pooled[b * 128 + d] = sum / (cnt + 1e-6f);
}

// ---------------- head: LN -> gelu(xW1+b1) -> xW2+b2 ----------------
__global__ __launch_bounds__(128) void k_head(const float* __restrict__ pooled,
                                              const float* __restrict__ g,
                                              const float* __restrict__ bb,
                                              const float* __restrict__ w1,
                                              const float* __restrict__ b1,
                                              const float* __restrict__ w2,
                                              const float* __restrict__ b2,
                                              float* __restrict__ out) {
    __shared__ float sv[128], su[128], red[128];
    int b = blockIdx.x, d = threadIdx.x;
    sv[d] = pooled[b * 128 + d];
    __syncthreads();
    red[d] = sv[d]; __syncthreads();
    for (int o = 64; o > 0; o >>= 1) { if (d < o) red[d] += red[d + o]; __syncthreads(); }
    float mean = red[0] * (1.0f / 128.0f);
    __syncthreads();
    float t0 = sv[d] - mean;
    red[d] = t0 * t0; __syncthreads();
    for (int o = 64; o > 0; o >>= 1) { if (d < o) red[d] += red[d + o]; __syncthreads(); }
    float var = red[0] * (1.0f / 128.0f);
    float rs = rsqrtf(var + 1e-5f);
    su[d] = (sv[d] - mean) * rs * g[d] + bb[d];
    __syncthreads();
    float z = b1[d];
    for (int k = 0; k < 128; ++k) z += su[k] * w1[k * 128 + d];
    float gz = gelu_exact(z);
    __syncthreads();
    red[d] = gz * w2[d]; __syncthreads();
    for (int o = 64; o > 0; o >>= 1) { if (d < o) red[d] += red[d + o]; __syncthreads(); }
    if (d == 0) out[b] = red[0] + b2[0];
}

// ---------------- launch ----------------
extern "C" void kernel_launch(void* const* d_in, const int* in_sizes, int n_in,
                              void* d_out, int out_size, void* d_ws, size_t ws_size,
                              hipStream_t stream) {
    (void)in_sizes; (void)n_in; (void)out_size; (void)ws_size;
    const float* seq       = (const float*)d_in[0];
    const unsigned char* mask = (const unsigned char*)d_in[1];   // jnp.bool_ -> 1 byte
    const float* embed_w   = (const float*)d_in[2];
    const float* embed_b   = (const float*)d_in[3];
    const float* ln1_g     = (const float*)d_in[4];
    const float* ln1_b     = (const float*)d_in[5];
    const float* decay     = (const float*)d_in[6];
    const float* proj_w    = (const float*)d_in[7];
    const float* proj_b    = (const float*)d_in[8];
    const float* dw_w      = (const float*)d_in[9];
    const float* dw_b      = (const float*)d_in[10];
    const float* ln2_g     = (const float*)d_in[11];
    const float* ln2_b     = (const float*)d_in[12];
    const float* mlp_w1    = (const float*)d_in[13];
    const float* mlp_b1    = (const float*)d_in[14];
    const float* mlp_w2    = (const float*)d_in[15];
    const float* mlp_b2    = (const float*)d_in[16];
    const float* head_ln_g = (const float*)d_in[17];
    const float* head_ln_b = (const float*)d_in[18];
    const float* head_w1   = (const float*)d_in[19];
    const float* head_b1   = (const float*)d_in[20];
    const float* head_w2   = (const float*)d_in[21];
    const float* head_b2   = (const float*)d_in[22];
    float* out = (float*)d_out;

    // workspace layout
    const size_t SZ_H = (size_t)TOK_ * D_ * sizeof(float);     // 64 MB
    char* w = (char*)d_ws;
    float* h    = (float*)(w);
    float* xn   = (float*)(w + SZ_H);
    char*  bigc = w + 2 * SZ_H;                                 // 128 MB region, reused:
    float*    emaf  = (float*)bigc;                             //  [0,64M)  EMA local states
    _Float16* xn16  = (_Float16*)bigc;                          //  [0,32M)  (after emaf dead)
    _Float16* ema16 = (_Float16*)(bigc + SZ_H);                 //  [64M,96M)
    _Float16* tmp16 = (_Float16*)(bigc + SZ_H);                 //  [64M,128M) (after ema16 dead)
    _Float16* projT = (_Float16*)(w + 4 * SZ_H);
    _Float16* w1T   = projT + (size_t)L_ * 128 * 128;
    _Float16* w2T   = w1T + (size_t)L_ * 256 * 128;
    float* carries  = (float*)(w + 4 * SZ_H + (1 << 20));
    float* Sin      = carries + (size_t)B_ * NC_ * D_;
    float* pooled   = Sin + (size_t)B_ * NC_ * D_;

    // weight prep (f16, transposed [N][K])
    for (int l = 0; l < L_; ++l) {
        k_transpose_f16<<<64, 256, 0, stream>>>(proj_w + (size_t)l * 128 * 128,
                                                projT + (size_t)l * 128 * 128, 128, 128);
        k_transpose_f16<<<128, 256, 0, stream>>>(mlp_w1 + (size_t)l * 128 * 256,
                                                 w1T + (size_t)l * 256 * 128, 128, 256);
        k_transpose_f16<<<128, 256, 0, stream>>>(mlp_w2 + (size_t)l * 256 * 128,
                                                 w2T + (size_t)l * 128 * 256, 256, 128);
    }

    k_embed<<<TOK_ * D_ / 256, 256, 0, stream>>>(seq, embed_w, embed_b, h);

    for (int l = 0; l < L_; ++l) {
        k_ln<false><<<TOK_ / 8, 256, 0, stream>>>(h, ln1_g + l * 128, ln1_b + l * 128,
                                                  xn, nullptr);
        k_ema_a<<<B_ * NC_ * D_ / 256, 256, 0, stream>>>(xn, decay + l * 128, emaf, carries);
        k_ema_b<<<B_ * D_ / 256, 256, 0, stream>>>(carries, decay + l * 128, Sin);
        k_ema_c<<<B_ * NC_ * D_ / 256, 256, 0, stream>>>(emaf, decay + l * 128, Sin, ema16);
        k_gemm<128, 128, 0><<<TOK_ / 64, 256, 0, stream>>>(
            ema16, projT + (size_t)l * 128 * 128, h, nullptr, proj_b + l * 128,
            xn, dw_w + l * 128 * 5, dw_b + l * 128);
        k_ln<true><<<TOK_ / 8, 256, 0, stream>>>(h, ln2_g + l * 128, ln2_b + l * 128,
                                                 xn, xn16);
        k_gemm<128, 256, 1><<<TOK_ / 64, 256, 0, stream>>>(
            xn16, w1T + (size_t)l * 256 * 128, nullptr, tmp16, mlp_b1 + l * 256,
            nullptr, nullptr, nullptr);
        k_gemm<256, 128, 2><<<TOK_ / 64, 256, 0, stream>>>(
            tmp16, w2T + (size_t)l * 128 * 256, h, nullptr, mlp_b2 + l * 128,
            nullptr, nullptr, nullptr);
    }

    k_pool<<<B_, 128, 0, stream>>>(h, mask, pooled);
    k_head<<<B_, 128, 0, stream>>>(pooled, head_ln_g, head_ln_b,
                                   head_w1, head_b1, head_w2, head_b2, out);
}
